// MagicPoint_77378130805032
// MI455X (gfx1250) — matrix operand order repaired and checked by reading, not compiled, not run on hardware
//
#include <hip/hip_runtime.h>
#include <hip/hip_bf16.h>

typedef __attribute__((ext_vector_type(16))) _Float16 v16h;
typedef __attribute__((ext_vector_type(8)))  _Float16 v8h;
typedef __attribute__((ext_vector_type(8)))  float    v8f;

union FragU { v16h v; v8h h[2]; };

#if defined(__has_builtin)
#if __has_builtin(__builtin_amdgcn_global_load_async_to_lds_b128)
#define ASYNC_LDS 1
#endif
#endif

#ifdef ASYNC_LDS
typedef int v4i_ __attribute__((vector_size(4 * sizeof(int))));
typedef __attribute__((address_space(1))) v4i_ GV4;   // global (printed as __device__)
typedef __attribute__((address_space(3))) v4i_ LV4;   // LDS (printed as __shared__)
#endif

// ---------------------------------------------------------------------------
// Weight packing: OIHW fp32 -> fp16 WMMA B-fragment order.
//   flat = ((nt*Kt32 + kt)*32 + lane)*16 + h
//   n = nt*16 + (lane&15),  k = kt*32 + (lane>>4)*16 + h,  k = tap*cin + c
// ---------------------------------------------------------------------------
__global__ void pack_weights_k(const float* __restrict__ w, _Float16* __restrict__ wp,
                               int cin, int cout, int cout_pad, int taps) {
    const int Ktot  = taps * cin;
    const int Kt32  = Ktot / 32;
    const int Nt    = cout_pad / 16;
    const unsigned total = (unsigned)Kt32 * Nt * 512u;
    for (unsigned i = blockIdx.x * blockDim.x + threadIdx.x; i < total;
         i += gridDim.x * blockDim.x) {
        const int h    = (int)(i & 15u);
        const int lane = (int)((i >> 4) & 31u);
        const unsigned tile = i >> 9;
        const int kt = (int)(tile % (unsigned)Kt32);
        const int nt = (int)(tile / (unsigned)Kt32);
        const int n  = nt * 16 + (lane & 15);
        const int k  = kt * 32 + ((lane >> 4) * 16) + h;
        float val = 0.0f;
        if (n < cout && k < Ktot) {
            const int tap = k / cin;
            const int c   = k - tap * cin;
            val = w[(unsigned)(n * cin + c) * (unsigned)taps + (unsigned)tap];
        }
        wp[i] = (_Float16)val;
    }
}

// ---------------------------------------------------------------------------
// b1a: cin==1 direct conv (3x3 SAME) + affine + ReLU -> fp16 NHWC(64)
// ---------------------------------------------------------------------------
__global__ __launch_bounds__(256)
void conv_b1a_k(const float* __restrict__ img, const float* __restrict__ w,
                const float* __restrict__ bb, const float* __restrict__ ss,
                const float* __restrict__ tt, _Float16* __restrict__ out,
                int Bn, int H, int W) {
    __shared__ float wsm[64 * 9];
    __shared__ float scl[64];
    __shared__ float shf[64];
    for (int i = threadIdx.x; i < 576; i += 256) wsm[i] = w[i];
    for (int i = threadIdx.x; i < 64; i += 256) {
        const float s = ss[i];
        scl[i] = s;
        shf[i] = bb[i] * s + tt[i];
    }
    __syncthreads();
    const unsigned HW  = (unsigned)H * W;
    const unsigned P   = (unsigned)Bn * HW;
    const unsigned pix = blockIdx.x * 256u + threadIdx.x;
    if (pix >= P) return;
    const unsigned b   = pix / HW;
    const unsigned rem = pix - b * HW;
    const int y = (int)(rem / (unsigned)W);
    const int x = (int)(rem - (unsigned)y * W);
    const float* ip = img + (size_t)b * HW;
    float v[9];
#pragma unroll
    for (int tap = 0; tap < 9; ++tap) {
        const int dy = tap / 3 - 1, dx = tap % 3 - 1;
        const int yy = y + dy, xx = x + dx;
        v[tap] = (yy >= 0 && yy < H && xx >= 0 && xx < W) ? ip[yy * W + xx] : 0.0f;
    }
    _Float16* o = out + (size_t)pix * 64;
#pragma unroll 4
    for (int c = 0; c < 64; ++c) {
        float acc = 0.0f;
#pragma unroll
        for (int tap = 0; tap < 9; ++tap) acc = fmaf(wsm[c * 9 + tap], v[tap], acc);
        o[c] = (_Float16)fmaxf(acc * scl[c] + shf[c], 0.0f);
    }
}

// ---------------------------------------------------------------------------
// Implicit-GEMM conv, v_wmma_f32_16x16x32_f16.
// Block = 8 waves; macro tile 64 pixels x 64 couts; each wave: 16px x 32couts
// (2 accumulators). A tile 64x64 halves (8KB LDS) staged per iteration
// (async-to-LDS when available); 4 WMMAs per barrier pair, A reused across N.
// cin % 64 == 0 for every layer; K order = tap-major, channel-minor.
// ---------------------------------------------------------------------------
__global__ __launch_bounds__(256)
void conv_wmma_k(_Float16* __restrict__ in, const _Float16* __restrict__ wpack,
                 const float* __restrict__ bp, const float* __restrict__ sp,
                 const float* __restrict__ tp, _Float16* __restrict__ out,
                 int Bn, int H, int W, int cin, int cout, int cout_stride,
                 int taps, int relu) {
    __shared__ _Float16 Asmem[64 * 64];

    const int tid  = threadIdx.x;
    const int lane = tid & 31;
    const int wave = tid >> 5;
    const int wm   = wave & 3;   // pixel sub-tile (16 rows)
    const int wn   = wave >> 2;  // cout pair sub-tile (2 x 16 cols)
    const unsigned pixbase = blockIdx.x * 64u;
    const int Kt32   = taps * cin / 32;
    const int Kt64   = Kt32 >> 1;
    const int ntbase = blockIdx.y * 4 + wn * 2;
    const unsigned HW = (unsigned)H * W;
    const unsigned P  = (unsigned)Bn * HW;

    // staging role: row = tid>>2 (0..63); two 8-half segments per thread
    const int srow = tid >> 2;
    const int seg0 = tid & 3;
    const unsigned spix = pixbase + (unsigned)srow;
    int sb = 0, sy = 0, sx = 0;
    const bool svalid = spix < P;
    if (svalid) {
        sb = (int)(spix / HW);
        const unsigned rem = spix - (unsigned)sb * HW;
        sy = (int)(rem / (unsigned)W);
        sx = (int)(rem - (unsigned)sy * W);
    }

    v8f acc0 = {}, acc1 = {};
    int tap = 0, cb = 0;  // incremental k decomposition (step 64 channels)
    for (int kt = 0; kt < Kt64; ++kt) {
        const int dy = (taps == 9) ? (tap / 3 - 1) : 0;
        const int dx = (taps == 9) ? (tap % 3 - 1) : 0;
        const int yy = sy + dy, xx = sx + dx;
        const bool vrow = svalid && yy >= 0 && yy < H && xx >= 0 && xx < W;
        _Float16* gbase = in + (size_t)((unsigned)(sb * H + yy) * (unsigned)W + (unsigned)xx) * cin + cb;
#pragma unroll
        for (int it = 0; it < 2; ++it) {
            const int seg = seg0 + it * 4;          // 8 segments of 8 halves
            _Float16* ldst = &Asmem[srow * 64 + seg * 8];
            if (vrow) {
#ifdef ASYNC_LDS
                __builtin_amdgcn_global_load_async_to_lds_b128(
                    (GV4*)(gbase + seg * 8), (LV4*)ldst, 0, 0);
#else
                *(float4*)ldst = *(const float4*)(gbase + seg * 8);
#endif
            } else {
                *(float4*)ldst = make_float4(0.f, 0.f, 0.f, 0.f);
            }
        }
#ifdef ASYNC_LDS
        asm volatile("s_wait_asynccnt 0" ::: "memory");
#endif
        __syncthreads();

        const int kglob2 = kt * 2;
#pragma unroll
        for (int kk = 0; kk < 2; ++kk) {
            // A fragment (contiguous k = 16*(lane>>4) + s, matching packed B): two ds_load_b128
            const _Float16* arow =
                &Asmem[(wm * 16 + (lane & 15)) * 64 + kk * 32 + ((lane >> 4) * 16)];
            FragU a;
            a.h[0] = *(const v8h*)(arow);
            a.h[1] = *(const v8h*)(arow + 8);

            const int ktg = kglob2 + kk;
#pragma unroll
            for (int nn = 0; nn < 2; ++nn) {
                const _Float16* bptr =
                    wpack + ((size_t)(ntbase + nn) * Kt32 + ktg) * 512 + lane * 16;
                FragU bf;
                bf.h[0] = *(const v8h*)(bptr);
                bf.h[1] = *(const v8h*)(bptr + 8);
                if (nn == 0)
                    acc0 = __builtin_amdgcn_wmma_f32_16x16x32_f16(
                        false, a.v, false, bf.v, (short)0, acc0, false, false);
                else
                    acc1 = __builtin_amdgcn_wmma_f32_16x16x32_f16(
                        false, a.v, false, bf.v, (short)0, acc1, false, false);
            }
        }
        if (kt + 1 < Kt64)
            __builtin_prefetch(
                wpack + ((size_t)ntbase * Kt32 + kglob2 + 2) * 512 + lane * 16, 0, 3);
        __syncthreads();

        cb += 64;
        if (cb >= cin) { cb = 0; ++tap; }
    }

    // epilogue: D element (m = (lane>>4)*8 + v, n = lane&15); fused affine+ReLU
    const int nl = lane & 15;
    const int mb = (lane >> 4) * 8;
#pragma unroll
    for (int nn = 0; nn < 2; ++nn) {
        const int cg = (ntbase + nn) * 16 + nl;
        float scl = 0.0f, shf = 0.0f;
        if (cg < cout) {
            const float s = sp[cg];
            scl = s;
            shf = bp[cg] * s + tp[cg];
        }
        const v8f acc = nn ? acc1 : acc0;
#pragma unroll
        for (int v = 0; v < 8; ++v) {
            const unsigned pix = pixbase + (unsigned)(wm * 16 + mb + v);
            if (pix < P && cg < cout_stride) {
                float y = acc[v] * scl + shf;
                if (relu) y = fmaxf(y, 0.0f);
                out[(size_t)pix * cout_stride + cg] = (_Float16)y;
            }
        }
    }
}

// ---------------------------------------------------------------------------
// 2x2 max pool, fp16 NHWC, 8 channels per thread (b128 loads)
// ---------------------------------------------------------------------------
__global__ void pool2_k(const _Float16* __restrict__ in, _Float16* __restrict__ out,
                        int Bn, int H, int W, int C) {
    const int Ho = H / 2, Wo = W / 2, CG = C / 8;
    const unsigned total = (unsigned)Bn * Ho * Wo * CG;
    const unsigned idx = blockIdx.x * blockDim.x + threadIdx.x;
    if (idx >= total) return;
    const int cg = (int)(idx % (unsigned)CG);
    unsigned pr = idx / (unsigned)CG;
    const int wo = (int)(pr % (unsigned)Wo); pr /= (unsigned)Wo;
    const int ho = (int)(pr % (unsigned)Ho);
    const int b  = (int)(pr / (unsigned)Ho);
    const _Float16* p00 = in + (size_t)((unsigned)(b * H + 2 * ho) * W + 2 * wo) * C + cg * 8;
    const _Float16* p10 = p00 + (size_t)W * C;
    const v8h a0 = *(const v8h*)p00;
    const v8h a1 = *(const v8h*)(p00 + C);
    const v8h a2 = *(const v8h*)p10;
    const v8h a3 = *(const v8h*)(p10 + C);
    v8h r;
#pragma unroll
    for (int i = 0; i < 8; ++i) {
        const float m = fmaxf(fmaxf((float)a0[i], (float)a1[i]),
                              fmaxf((float)a2[i], (float)a3[i]));
        r[i] = (_Float16)m;
    }
    *(v8h*)(out + (size_t)idx * 8) = r;
}

// ---------------------------------------------------------------------------
// 65-way softmax (drop dustbin) + depth_to_space(8) scatter -> fp32 score map
// ---------------------------------------------------------------------------
__global__ void softmax_shuffle_k(const _Float16* __restrict__ logits,
                                  float* __restrict__ score,
                                  int Bn, int Hc, int Wc) {
    const unsigned P   = (unsigned)Bn * Hc * Wc;
    const unsigned pix = blockIdx.x * blockDim.x + threadIdx.x;
    if (pix >= P) return;
    const _Float16* l = logits + (size_t)pix * 96;
    float mx = -3.0e38f;
#pragma unroll
    for (int c = 0; c < 65; ++c) mx = fmaxf(mx, (float)l[c]);
    float e[64];
    float sum = __expf((float)l[64] - mx);  // dustbin
#pragma unroll
    for (int c = 0; c < 64; ++c) {
        const float ev = __expf((float)l[c] - mx);
        e[c] = ev;
        sum += ev;
    }
    const float inv = 1.0f / sum;
    const int wc = (int)(pix % (unsigned)Wc);
    unsigned pr = pix / (unsigned)Wc;
    const int hc = (int)(pr % (unsigned)Hc);
    const int b  = (int)(pr / (unsigned)Hc);
    const int H = Hc * 8, W = Wc * 8;
    float* sb = score + (size_t)b * H * W;
#pragma unroll
    for (int i = 0; i < 8; ++i) {
        float* rowp = sb + (size_t)(hc * 8 + i) * W + wc * 8;
#pragma unroll
        for (int j = 0; j < 8; ++j) rowp[j] = e[i * 8 + j] * inv;
    }
}

// ---------------------------------------------------------------------------
// Separable 9x9 max (radius 4, clamp == -inf pad), compare+border+threshold
// ---------------------------------------------------------------------------
__global__ void nms_row_k(const float* __restrict__ s, float* __restrict__ r,
                          int Bn, int H, int W) {
    const unsigned total = (unsigned)Bn * H * W;
    const unsigned idx = blockIdx.x * blockDim.x + threadIdx.x;
    if (idx >= total) return;
    const int x = (int)(idx % (unsigned)W);
    const float* sr = s + (idx - (unsigned)x);
    const int x0 = max(x - 4, 0), x1 = min(x + 4, W - 1);
    float m = -3.0e38f;
    for (int xx = x0; xx <= x1; ++xx) m = fmaxf(m, sr[xx]);
    r[idx] = m;
}

__global__ void nms_final_k(const float* __restrict__ s, const float* __restrict__ r,
                            float* __restrict__ out, int Bn, int H, int W) {
    const unsigned total = (unsigned)Bn * H * W;
    const unsigned idx = blockIdx.x * blockDim.x + threadIdx.x;
    if (idx >= total) return;
    const int x = (int)(idx % (unsigned)W);
    unsigned pr = idx / (unsigned)W;
    const int y = (int)(pr % (unsigned)H);
    const int b = (int)(pr / (unsigned)H);
    const int y0 = max(y - 4, 0), y1 = min(y + 4, H - 1);
    const float* rb = r + (size_t)b * H * W + x;
    float m = -3.0e38f;
    for (int yy = y0; yy <= y1; ++yy) m = fmaxf(m, rb[(size_t)yy * W]);
    const float v = s[idx];
    const bool keep = (v == m) && (y >= 4) && (y < H - 4) && (x >= 4) && (x < W - 4)
                      && (v >= 0.015f);
    out[idx] = keep ? v : 0.0f;
}

// ---------------------------------------------------------------------------
// Host orchestration
// ---------------------------------------------------------------------------
static inline size_t packed_halves(int cin, int cout_pad, int taps) {
    return (size_t)((taps * cin) / 32) * (cout_pad / 16) * 512;
}

extern "C" void kernel_launch(void* const* d_in, const int* in_sizes, int n_in,
                              void* d_out, int out_size, void* d_ws, size_t ws_size,
                              hipStream_t stream) {
    (void)in_sizes; (void)n_in; (void)out_size; (void)ws_size;
    const int Bn = 8, H0 = 480, W0 = 640;

    char* wsp = (char*)d_ws;
    auto carve = [&](size_t bytes) {
        void* r = (void*)wsp;
        wsp += (bytes + 255) & ~(size_t)255;
        return r;
    };
    const size_t actBytes = (size_t)Bn * H0 * W0 * 64 * sizeof(_Float16);  // 315 MB
    _Float16* actA  = (_Float16*)carve(actBytes);
    _Float16* actB  = (_Float16*)carve(actBytes);
    _Float16* actH2 = (_Float16*)carve((size_t)Bn * 60 * 80 * 96 * sizeof(_Float16));
    float*    score = (float*)carve((size_t)Bn * H0 * W0 * sizeof(float));
    float*    rowm  = (float*)carve((size_t)Bn * H0 * W0 * sizeof(float));

    struct L { int l, cin, cout, cout_pad, taps; };
    const L LS[9] = {
        {1,  64,  64,  64, 9},  // b1b
        {2,  64,  64,  64, 9},  // b2a
        {3,  64,  64,  64, 9},  // b2b
        {4,  64, 128, 128, 9},  // b3a
        {5, 128, 128, 128, 9},  // b3b
        {6, 128, 128, 128, 9},  // b4a
        {7, 128, 128, 128, 9},  // b4b
        {8, 128, 256, 256, 9},  // h1
        {9, 256,  65, 128, 1},  // h2 (padded to 128 for the 64-wide N macro tile)
    };
    _Float16* wp[9];
    for (int i = 0; i < 9; ++i) {
        const size_t ph = packed_halves(LS[i].cin, LS[i].cout_pad, LS[i].taps);
        wp[i] = (_Float16*)carve(ph * sizeof(_Float16));
        const int blocks = (int)((ph + 255) / 256);
        pack_weights_k<<<blocks, 256, 0, stream>>>(
            (const float*)d_in[1 + 4 * LS[i].l], wp[i],
            LS[i].cin, LS[i].cout, LS[i].cout_pad, LS[i].taps);
    }

    auto conv = [&](int i, _Float16* in, _Float16* out, int H, int W,
                    int cout_stride, int relu) {
        const L& c = LS[i];
        const unsigned P = (unsigned)Bn * H * W;
        dim3 g((P + 63) / 64, (unsigned)(c.cout_pad / 64));
        const int base = 1 + 4 * c.l;
        conv_wmma_k<<<g, 256, 0, stream>>>(
            in, wp[i],
            (const float*)d_in[base + 1], (const float*)d_in[base + 2],
            (const float*)d_in[base + 3], out,
            Bn, H, W, c.cin, c.cout, cout_stride, c.taps, relu);
    };
    auto pool = [&](const _Float16* in, _Float16* out, int H, int W, int C) {
        const unsigned total = (unsigned)Bn * (H / 2) * (W / 2) * (C / 8);
        pool2_k<<<(total + 255) / 256, 256, 0, stream>>>(in, out, Bn, H, W, C);
    };

    {
        const unsigned P = (unsigned)Bn * H0 * W0;
        conv_b1a_k<<<(P + 255) / 256, 256, 0, stream>>>(
            (const float*)d_in[0], (const float*)d_in[1], (const float*)d_in[2],
            (const float*)d_in[3], (const float*)d_in[4], actA, Bn, H0, W0);
    }
    conv(0, actA, actB, 480, 640, 64, 1);          // b1b
    pool(actB, actA, 480, 640, 64);
    conv(1, actA, actB, 240, 320, 64, 1);          // b2a
    conv(2, actB, actA, 240, 320, 64, 1);          // b2b
    pool(actA, actB, 240, 320, 64);
    conv(3, actB, actA, 120, 160, 128, 1);         // b3a
    conv(4, actA, actB, 120, 160, 128, 1);         // b3b
    pool(actB, actA, 120, 160, 128);
    conv(5, actA, actB, 60, 80, 128, 1);           // b4a
    conv(6, actB, actA, 60, 80, 128, 1);           // b4b
    conv(7, actA, actB, 60, 80, 256, 1);           // h1
    conv(8, actB, actH2, 60, 80, 96, 0);           // h2 (no ReLU, stride 96)

    {
        const unsigned P = (unsigned)Bn * 60 * 80;
        softmax_shuffle_k<<<(P + 255) / 256, 256, 0, stream>>>(actH2, score, Bn, 60, 80);
    }
    {
        const unsigned total = (unsigned)Bn * H0 * W0;
        nms_row_k<<<(total + 255) / 256, 256, 0, stream>>>(score, rowm, Bn, H0, W0);
        nms_final_k<<<(total + 255) / 256, 256, 0, stream>>>(
            score, rowm, (float*)d_out, Bn, H0, W0);
    }
}
